// SSIMLoss_81518479278455
// MI455X (gfx1250) — matrix operand order repaired and checked by reading, not compiled, not run on hardware
//
#include <hip/hip_runtime.h>
#include <hip/hip_bf16.h>

typedef __attribute__((ext_vector_type(16))) __bf16 v16bf;
typedef __attribute__((ext_vector_type(8)))  float  v8f;

union V16U {
    v16bf v;
    int4  q4[2];
    unsigned int ui[8];
    unsigned short u[16];
};

union PK8U {
    int4 q4;
    unsigned int ui[4];
};

__device__ __forceinline__ unsigned short f32_to_bf16_bits(float x) {
    union { float f; unsigned int u; } a;
    a.f = x;
    unsigned int b = a.u;
    unsigned int r = b + 0x7FFFu + ((b >> 16) & 1u);   // RNE (one-time setup only)
    return (unsigned short)(r >> 16);
}

// Pack two f32 -> two bf16 (round-to-nearest) in 3 VALU ops:
// two v_add_nc_u32 (+0x8000 bias) + one v_perm_b32 picking the high halves.
__device__ __forceinline__ unsigned int cvt2_bf16(float a, float b) {
    union { float f; unsigned int u; } x, y;
    x.f = a; y.f = b;
    unsigned int ra = x.u + 0x8000u;
    unsigned int rb = y.u + 0x8000u;
    return __builtin_amdgcn_perm(rb, ra, 0x07060302u);
}

#define IMG_W 256
#define IMG_PIX 65536            // 256*256
#define N_PIX 12582912.0f        // 64*3*256*256
#define MIDSTRIDE 48             // rows per column in mid staging (48*2B = 96B, 16B-aligned strides)

__global__ __launch_bounds__(256)
void ssim_main_kernel(const float* __restrict__ pred,
                      const float* __restrict__ targ,
                      float* __restrict__ accum)
{
    // Per-wave column-major bf16 staging of horizontal-conv results:
    // mid[wave][q][col*48 + row], 16 cols x 48 rows.  8*5*768*2B = 60 KB.
    __shared__ alignas(16) unsigned short mid[8][5][16 * MIDSTRIDE];

    const int lane = threadIdx.x & 31;
    const int wave = threadIdx.x >> 5;
    const int lo   = lane & 15;
    const int hi   = lane >> 4;

    // Tile = 32 output rows x 16 output cols.  192 planes * 8 * 16 = 24576 tiles.
    const int tile = blockIdx.x * 8 + wave;
    const int img  = tile >> 7;                  // 128 tiles per 256x256 plane
    const int r0   = ((tile >> 4) & 7) * 32;
    const int c0   = (tile & 15) * 16;
    const float* __restrict__ pb = pred + (size_t)img * IMG_PIX;
    const float* __restrict__ tb = targ + (size_t)img * IMG_PIX;

    // Wave-uniform: input window rows r0-5..r0+42, cols c0-5..c0+26 fully in-image?
    const bool interior = (r0 >= 5) && (r0 + 42 < 256) && (c0 >= 5) && (c0 + 26 < 256);

    // ---- Gaussian taps: g[j] = exp(-(j-5)^2/4.5)/sum, j in [0,10] ----
    const float INVS = 0.26601178f;   // 1 / sum(exp(-d^2/4.5), d=-5..5)

    // Horizontal band matrix as WMMA B operand (32x16):
    // B[k][c] = g[k-c] when 0<=k-c<=10.  Lane n: col=n&15, K = 8*(n>>4) + (e<8?e:e+8).
    V16U bandB;
#pragma unroll
    for (int e = 0; e < 16; ++e) {
        int K = hi * 8 + (e < 8 ? e : e + 8);
        int j = K - lo;
        float d = (float)(j - 5);
        float w = __expf(d * d * (-1.0f / 4.5f)) * INVS;
        bandB.u[e] = (j >= 0 && j <= 10) ? f32_to_bf16_bits(w) : (unsigned short)0;
    }

    // Vertical band matrix as WMMA A operand (16x32):
    // A[m][k] = g[k-m] when 0<=k-m<=10.  Lane: M=lo, K = hi*16 + e.
    V16U bandA;
#pragma unroll
    for (int e = 0; e < 16; ++e) {
        int K = hi * 16 + e;
        int j = K - lo;
        float d = (float)(j - 5);
        float w = __expf(d * d * (-1.0f / 4.5f)) * INVS;
        bandA.u[e] = (j >= 0 && j <= 10) ? f32_to_bf16_bits(w) : (unsigned short)0;
    }

    // ---- Horizontal pass: three 16-row halves covering input rows r0-5 .. r0+42 ----
#pragma unroll
    for (int h = 0; h < 3; ++h) {
        const int gr = r0 - 5 + h * 16 + lo;     // A operand: row = lo

        float fp[16], ft[16];
        if (interior) {
            // Wave-uniform fast path: no padding selects at all.
            const float* __restrict__ prow = pb + gr * IMG_W + (c0 - 5);
            const float* __restrict__ trow = tb + gr * IMG_W + (c0 - 5);
#pragma unroll
            for (int e = 0; e < 16; ++e) {
                int K = hi * 8 + (e < 8 ? e : e + 8);  // A layout K index
                fp[e] = prow[K];
                ft[e] = trow[K];
            }
        } else {
            const bool rok = ((unsigned)gr < 256u);
#pragma unroll
            for (int e = 0; e < 16; ++e) {
                int K  = hi * 8 + (e < 8 ? e : e + 8);
                int gc = c0 - 5 + K;
                bool ok = rok && ((unsigned)gc < 256u);
                int idx = ok ? (gr * IMG_W + gc) : 0;   // branchless, always-valid addr
                float pv = pb[idx];
                float tv = tb[idx];
                fp[e] = ok ? pv : 0.0f;                 // zero padding (matches reference)
                ft[e] = ok ? tv : 0.0f;
            }
        }

#pragma unroll
        for (int q = 0; q < 5; ++q) {
            V16U A;
#pragma unroll
            for (int e = 0; e < 16; e += 2) {
                float v0, v1;
                if      (q == 0) { v0 = fp[e];         v1 = fp[e + 1]; }
                else if (q == 1) { v0 = ft[e];         v1 = ft[e + 1]; }
                else if (q == 2) { v0 = fp[e] * fp[e]; v1 = fp[e + 1] * fp[e + 1]; }
                else if (q == 3) { v0 = ft[e] * ft[e]; v1 = ft[e + 1] * ft[e + 1]; }
                else             { v0 = fp[e] * ft[e]; v1 = fp[e + 1] * ft[e + 1]; }
                A.ui[e >> 1] = cvt2_bf16(v0, v1);
            }
            v8f cz = {};
            v8f d = __builtin_amdgcn_wmma_f32_16x16x32_bf16(
                false, A.v, false, bandB.v, (short)0, cz, false, false);

            // C layout: lane n holds rows 8*(n>>4)+i (i=0..7), col n&15.
            // Pack 8 consecutive rows -> column-major LDS, one 16B store.
            PK8U pk;
#pragma unroll
            for (int i = 0; i < 4; ++i)
                pk.ui[i] = cvt2_bf16(d[2 * i], d[2 * i + 1]);
            *(int4*)&mid[wave][q][lo * MIDSTRIDE + h * 16 + hi * 8] = pk.q4;
        }
    }

    // ---- Vertical pass + SSIM: two 16-row output sub-tiles ----
    const float C1 = 1e-4f;   // 0.01^2
    const float C2 = 9e-4f;   // 0.03^2
    float s = 0.0f;
#pragma unroll
    for (int t = 0; t < 2; ++t) {
        v8f outq[5];
#pragma unroll
        for (int q = 0; q < 5; ++q) {
            V16U B;   // lane n: col n&15, 16 consecutive mid rows from 16t + 16*(n>>4)
            const unsigned short* src = &mid[wave][q][lo * MIDSTRIDE + t * 16 + hi * 16];
            B.q4[0] = *(const int4*)(src);
            B.q4[1] = *(const int4*)(src + 8);
            v8f cz = {};
            outq[q] = __builtin_amdgcn_wmma_f32_16x16x32_bf16(
                false, bandA.v, false, B.v, (short)0, cz, false, false);
        }
#pragma unroll
        for (int i = 0; i < 8; ++i) {
            float mu1 = outq[0][i];
            float mu2 = outq[1][i];
            float mu1s = mu1 * mu1;
            float mu2s = mu2 * mu2;
            float m12  = mu1 * mu2;
            float s1   = outq[2][i] - mu1s;
            float s2   = outq[3][i] - mu2s;
            float s12  = outq[4][i] - m12;
            float num  = (2.0f * m12 + C1) * (2.0f * s12 + C2);
            float den  = (mu1s + mu2s + C1) * (s1 + s2 + C2);
            s += num * __builtin_amdgcn_rcpf(den);   // v_rcp_f32: fine for a loss scalar
        }
    }

#pragma unroll
    for (int off = 16; off > 0; off >>= 1)
        s += __shfl_xor(s, off, 32);
    if (lane == 0)
        atomicAdd(accum, s);
}

__global__ void ssim_finalize_kernel(const float* __restrict__ accum,
                                     float* __restrict__ out)
{
    out[0] = 1.0f - accum[0] * (1.0f / N_PIX);
}

extern "C" void kernel_launch(void* const* d_in, const int* in_sizes, int n_in,
                              void* d_out, int out_size, void* d_ws, size_t ws_size,
                              hipStream_t stream) {
    const float* pred = (const float*)d_in[0];
    const float* targ = (const float*)d_in[1];
    // d_in[2] is the Gaussian window; we regenerate it analytically in-kernel.
    float* out = (float*)d_out;
    float* acc = (float*)d_ws;

    hipMemsetAsync(acc, 0, sizeof(float), stream);
    // 192 planes * 128 tiles = 24576 wave-tiles, 8 waves per 256-thread block
    ssim_main_kernel<<<3072, 256, 0, stream>>>(pred, targ, acc);
    ssim_finalize_kernel<<<1, 1, 0, stream>>>(acc, out);
}
